// DualAddressingPhasor_37082747633930
// MI455X (gfx1250) — matrix operand order repaired and checked by reading, not compiled, not run on hardware
//
#include <hip/hip_runtime.h>
#include <hip/hip_bf16.h>

#define Dd 512
#define Kk 32
#define Bb 2
#define Ll 2048
#define Mm (Bb * Ll)   // 4096 rows

typedef __attribute__((ext_vector_type(16))) __bf16 bf16x16;
typedef __attribute__((ext_vector_type(8)))  float  v8f;

struct alignas(16) U128 { unsigned int x[4]; };
union ABVec { bf16x16 v; U128 q[2]; };

// ---------- helpers ----------
__device__ __forceinline__ unsigned short f2bf(float f) {
    unsigned u = __float_as_uint(f);
    unsigned r = u + 0x7FFFu + ((u >> 16) & 1u);   // round-to-nearest-even
    return (unsigned short)(r >> 16);
}

// WMMA mainloop, register-blocked over N:
//   one A fragment (16x32 bf16) reused against NT B fragments -> NT accumulators.
// A lane layout (16-bit A 16x32): elem j -> K = 8*half + j (j<8), 16 + 8*half + (j-8)
// B lane layout (16-bit B 32x16): lane(16*half+n) holds col n, elem j -> K = 8*half + j (j<8), 16 + 8*half + (j-8)
template <int NT>
__device__ __forceinline__ void wmma_tiles(const unsigned short* __restrict__ A,
                                           const unsigned short* __restrict__ Bt,
                                           int lda, int ldb, int kdim,
                                           int row0, int col0, v8f acc[NT]) {
    const int lane = threadIdx.x & 31;
    const int half = lane >> 4;
    const int mn   = lane & 15;
    const unsigned short* arow = A + (size_t)(row0 + mn) * lda;
    const unsigned short* brow[NT];
#pragma unroll
    for (int t = 0; t < NT; ++t) {
        brow[t] = Bt + (size_t)(col0 + t * 16 + mn) * ldb;
        acc[t] = {};
    }
    for (int kk = 0; kk < kdim; kk += 32) {
        ABVec a;
        a.q[0] = *(const U128*)(arow + kk + 8 * half);
        a.q[1] = *(const U128*)(arow + kk + 16 + 8 * half);
        if (kk + 64 < kdim) __builtin_prefetch(arow + kk + 64, 0, 3);
#pragma unroll
        for (int t = 0; t < NT; ++t) {
            ABVec b;
            b.q[0] = *(const U128*)(brow[t] + kk + 8 * half);
            b.q[1] = *(const U128*)(brow[t] + kk + 16 + 8 * half);
            if (kk + 64 < kdim) __builtin_prefetch(brow[t] + kk + 64, 0, 3);
            acc[t] = __builtin_amdgcn_wmma_f32_16x16x32_bf16(
                false, a.v, false, b.v, (short)0, acc[t], false, false);
        }
    }
}

// ---------- prep ----------
__global__ void k_cvt_bf16(const float* __restrict__ in, unsigned short* __restrict__ out, int n) {
    int i = blockIdx.x * blockDim.x + threadIdx.x;
    if (i < n) out[i] = f2bf(in[i]);
}

// W[k][n] (rows x cols, f32) -> Wt[n][k] (bf16)
__global__ void k_transpose_bf16(const float* __restrict__ W, unsigned short* __restrict__ Wt,
                                 int rows, int cols) {
    int i = blockIdx.x * blockDim.x + threadIdx.x;
    if (i < rows * cols) {
        int n = i / rows, k = i % rows;
        Wt[(size_t)n * rows + k] = f2bf(W[(size_t)k * cols + n]);
    }
}

// ---------- GEMM 1: h = tanh(x@W1 + b1) -> bf16 ----------
__global__ void k_gemm_h(const unsigned short* __restrict__ xb,
                         const unsigned short* __restrict__ w1t,
                         const float* __restrict__ b1,
                         unsigned short* __restrict__ hb) {
    int wave = threadIdx.x >> 5;
    int row0 = blockIdx.x * 16;
    int col0 = (blockIdx.y * 2 + wave) * 64;
    v8f acc[4];
    wmma_tiles<4>(xb, w1t, Dd, Dd, Dd, row0, col0, acc);
    int lane = threadIdx.x & 31, half = lane >> 4;
#pragma unroll
    for (int t = 0; t < 4; ++t) {
        int n = col0 + t * 16 + (lane & 15);
        float bn = b1[n];
#pragma unroll
        for (int r = 0; r < 8; ++r) {
            int row = row0 + r + 8 * half;
            hb[(size_t)row * Dd + n] = f2bf(tanhf(acc[t][r] + bn));
        }
    }
}

// ---------- GEMM 2: value = x@Wv + bv -> f32 ----------
__global__ void k_gemm_v(const unsigned short* __restrict__ xb,
                         const unsigned short* __restrict__ wvt,
                         const float* __restrict__ bv,
                         float* __restrict__ val) {
    int wave = threadIdx.x >> 5;
    int row0 = blockIdx.x * 16;
    int col0 = (blockIdx.y * 2 + wave) * 64;
    v8f acc[4];
    wmma_tiles<4>(xb, wvt, Dd, Dd, Dd, row0, col0, acc);
    int lane = threadIdx.x & 31, half = lane >> 4;
#pragma unroll
    for (int t = 0; t < 4; ++t) {
        int n = col0 + t * 16 + (lane & 15);
        float bn = bv[n];
#pragma unroll
        for (int r = 0; r < 8; ++r) {
            int row = row0 + r + 8 * half;
            val[(size_t)row * Dd + n] = acc[t][r] + bn;
        }
    }
}

// ---------- GEMM 3: phases -> c,s [M,K] ----------
__global__ void k_gemm_phase(const unsigned short* __restrict__ hb,
                             const unsigned short* __restrict__ w2t,
                             const float* __restrict__ b2,
                             const float* __restrict__ pos_scale,
                             const float* __restrict__ content_scale,
                             float* __restrict__ cph, float* __restrict__ sph) {
    int wave = threadIdx.x >> 5;
    int row0 = blockIdx.x * 16;
    int col0 = wave * 16;                      // N = 32 -> 2 col tiles
    v8f acc[1];
    wmma_tiles<1>(hb, w2t, Dd, Dd, Dd, row0, col0, acc);
    int lane = threadIdx.x & 31, half = lane >> 4, k = col0 + (lane & 15);
    float ps = pos_scale[0], cs = content_scale[0];
    float bn = b2[k];
    float freq = __expf(-(float)k * (9.210340371976184f / 32.f)); // 10000^(-k/32)
#pragma unroll
    for (int r = 0; r < 8; ++r) {
        int row = row0 + r + 8 * half;
        int l = row & (Ll - 1);
        float cp = tanhf(acc[0][r] + bn) * 3.14159265358979f;
        float ph = ps * (float)l * freq + cs * cp;
        cph[(size_t)row * Kk + k] = cosf(ph);
        sph[(size_t)row * Kk + k] = sinf(ph);
    }
}

// ---------- scan: cumsum bind + retrieve (state in VGPRs, c/s staged in LDS) ----------
__global__ void __launch_bounds__(512)
k_scan(const float* __restrict__ val, const float* __restrict__ cph,
       const float* __restrict__ sph, float* __restrict__ retr) {
    const int b = blockIdx.x;
    const int d = threadIdx.x;          // one thread per feature column
    __shared__ float sc[64 * Kk];
    __shared__ float ss[64 * Kk];
    float Mr[Kk], Mi[Kk];
#pragma unroll
    for (int k = 0; k < Kk; ++k) { Mr[k] = 0.f; Mi[k] = 0.f; }

    for (int l0 = 0; l0 < Ll; l0 += 64) {
        __syncthreads();
#pragma unroll
        for (int t = 0; t < 4; ++t) {
            int idx = threadIdx.x + 512 * t;  // 2048 = 64*32
            sc[idx] = cph[((size_t)(b * Ll + l0)) * Kk + idx];
            ss[idx] = sph[((size_t)(b * Ll + l0)) * Kk + idx];
        }
        __syncthreads();
        for (int li = 0; li < 64; ++li) {
            int l = l0 + li;
            float v = val[((size_t)(b * Ll + l)) * Dd + d];
            float a = 0.f;
#pragma unroll
            for (int k = 0; k < Kk; ++k) {
                float cc = sc[li * Kk + k];
                float s2 = ss[li * Kk + k];
                Mr[k] = fmaf(cc, v, Mr[k]);
                Mi[k] = fmaf(s2, v, Mi[k]);
                a = fmaf(cc, Mr[k], a);
                a = fmaf(s2, Mi[k], a);
            }
            retr[((size_t)(b * Ll + l)) * Dd + d] = a * rsqrtf((float)(l + 1) * (float)Kk);
        }
    }
}

// ---------- LayerNorm: one wave32 per row -> bf16 ----------
__global__ void __launch_bounds__(256)
k_ln(const float* __restrict__ retr, const float* __restrict__ g,
     const float* __restrict__ be, unsigned short* __restrict__ lnb) {
    int row = blockIdx.x * 8 + (threadIdx.x >> 5);
    int lane = threadIdx.x & 31;
    float v[16], s1 = 0.f, s2 = 0.f;
#pragma unroll
    for (int j = 0; j < 16; ++j) {
        v[j] = retr[(size_t)row * Dd + lane + 32 * j];
        s1 += v[j];
        s2 += v[j] * v[j];
    }
#pragma unroll
    for (int off = 16; off > 0; off >>= 1) {
        s1 += __shfl_xor(s1, off, 32);
        s2 += __shfl_xor(s2, off, 32);
    }
    float mu = s1 * (1.f / Dd);
    float var = s2 * (1.f / Dd) - mu * mu;
    float rstd = rsqrtf(var + 1e-5f);
#pragma unroll
    for (int j = 0; j < 16; ++j) {
        int dcol = lane + 32 * j;
        lnb[(size_t)row * Dd + dcol] = f2bf((v[j] - mu) * rstd * g[dcol] + be[dcol]);
    }
}

// ---------- GEMM 4: out = x + ln@Wo + bo ----------
__global__ void k_gemm_out(const unsigned short* __restrict__ lnb,
                           const unsigned short* __restrict__ wot,
                           const float* __restrict__ bo,
                           const float* __restrict__ x,
                           float* __restrict__ out) {
    int wave = threadIdx.x >> 5;
    int row0 = blockIdx.x * 16;
    int col0 = (blockIdx.y * 2 + wave) * 64;
    v8f acc[4];
    wmma_tiles<4>(lnb, wot, Dd, Dd, Dd, row0, col0, acc);
    int lane = threadIdx.x & 31, half = lane >> 4;
#pragma unroll
    for (int t = 0; t < 4; ++t) {
        int n = col0 + t * 16 + (lane & 15);
        float bn = bo[n];
#pragma unroll
        for (int r = 0; r < 8; ++r) {
            int row = row0 + r + 8 * half;
            out[(size_t)row * Dd + n] = x[(size_t)row * Dd + n] + acc[t][r] + bn;
        }
    }
}

extern "C" void kernel_launch(void* const* d_in, const int* in_sizes, int n_in,
                              void* d_out, int out_size, void* d_ws, size_t ws_size,
                              hipStream_t stream) {
    const float* x  = (const float*)d_in[0];
    const float* W1 = (const float*)d_in[1];
    const float* b1 = (const float*)d_in[2];
    const float* W2 = (const float*)d_in[3];
    const float* b2 = (const float*)d_in[4];
    const float* ps = (const float*)d_in[5];
    const float* cs = (const float*)d_in[6];
    const float* Wv = (const float*)d_in[7];
    const float* bv = (const float*)d_in[8];
    const float* lg = (const float*)d_in[9];
    const float* lb = (const float*)d_in[10];
    const float* Wo = (const float*)d_in[11];
    const float* bo = (const float*)d_in[12];
    float* out = (float*)d_out;

    // workspace layout
    char* base = (char*)d_ws;
    size_t off = 0;
    auto take = [&](size_t bytes) { char* p = base + off; off = (off + bytes + 255) & ~(size_t)255; return p; };
    unsigned short* xb  = (unsigned short*)take((size_t)Mm * Dd * 2);
    unsigned short* hb  = (unsigned short*)take((size_t)Mm * Dd * 2);
    unsigned short* lnb = (unsigned short*)take((size_t)Mm * Dd * 2);
    unsigned short* w1t = (unsigned short*)take((size_t)Dd * Dd * 2);
    unsigned short* wvt = (unsigned short*)take((size_t)Dd * Dd * 2);
    unsigned short* wot = (unsigned short*)take((size_t)Dd * Dd * 2);
    unsigned short* w2t = (unsigned short*)take((size_t)Dd * Kk * 2);
    float* cph  = (float*)take((size_t)Mm * Kk * 4);
    float* sph  = (float*)take((size_t)Mm * Kk * 4);
    float* val  = (float*)take((size_t)Mm * Dd * 4);
    float* retr = (float*)take((size_t)Mm * Dd * 4);
    (void)ws_size; (void)n_in; (void)in_sizes; (void)out_size;

    // prep: conversions / transposes
    k_cvt_bf16<<<(Mm * Dd + 255) / 256, 256, 0, stream>>>(x, xb, Mm * Dd);
    k_transpose_bf16<<<(Dd * Dd + 255) / 256, 256, 0, stream>>>(W1, w1t, Dd, Dd);
    k_transpose_bf16<<<(Dd * Dd + 255) / 256, 256, 0, stream>>>(Wv, wvt, Dd, Dd);
    k_transpose_bf16<<<(Dd * Dd + 255) / 256, 256, 0, stream>>>(Wo, wot, Dd, Dd);
    k_transpose_bf16<<<(Dd * Kk + 255) / 256, 256, 0, stream>>>(W2, w2t, Dd, Kk);

    dim3 gfull(Mm / 16, Dd / 128);    // 256 x 4, 2 waves/block, each wave does 16x64
    k_gemm_h<<<gfull, 64, 0, stream>>>(xb, w1t, b1, hb);
    k_gemm_v<<<gfull, 64, 0, stream>>>(xb, wvt, bv, val);
    k_gemm_phase<<<dim3(Mm / 16, 1), 64, 0, stream>>>(hb, w2t, b2, ps, cs, cph, sph);
    k_scan<<<Bb, 512, 0, stream>>>(val, cph, sph, retr);
    k_ln<<<Mm / 8, 256, 0, stream>>>(retr, lg, lb, lnb);
    k_gemm_out<<<gfull, 64, 0, stream>>>(lnb, wot, bo, x, out);
}
